// Model_13365938225769
// MI455X (gfx1250) — matrix operand order repaired and checked by reading, not compiled, not run on hardware
//
#include <hip/hip_runtime.h>
#include <stdint.h>

// ---------------- types ----------------
typedef __bf16 bf16;
typedef __attribute__((ext_vector_type(16))) __bf16 v16bf;
typedef __attribute__((ext_vector_type(8)))  __bf16 bf16x8;
typedef __attribute__((ext_vector_type(8)))  float  v8f;
typedef __attribute__((ext_vector_type(4)))  int    v4i;

// ---------------- problem dims ----------------
#define B_    256
#define L_    96
#define CIN_  321
#define D_    512
#define H_    512
#define TOUT_ 192
#define PRED_ 96
#define NL_   2
#define NMARK_ 4
#define XPL_  98              // circular-padded time length
#define KCONV_ 992            // 3*321=963 padded up to multiple of 32

// ---------------- gfx1250 async LDS staging ----------------
#if defined(__has_builtin)
#if __has_builtin(__builtin_amdgcn_global_load_async_to_lds_b128) && __has_builtin(__builtin_amdgcn_s_wait_asynccnt)
#define USE_ASYNC_LDS 1
#endif
#endif

typedef __attribute__((address_space(3))) v4i* lds_v4i_p;

__device__ __forceinline__ void async_cp16(const void* g, void* l) {
#ifdef USE_ASYNC_LDS
  __builtin_amdgcn_global_load_async_to_lds_b128((v4i*)g, (lds_v4i_p)l, 0, 0);
#else
  *(uint4*)l = *(const uint4*)g;
#endif
}
__device__ __forceinline__ void wait_stage_barrier() {
#ifdef USE_ASYNC_LDS
  __builtin_amdgcn_s_wait_asynccnt(0);
#endif
  __syncthreads();
}

__device__ __forceinline__ float sigmoidf_(float x) { return 1.f / (1.f + expf(-x)); }
__device__ __forceinline__ float geluf_(float x)    { return 0.5f * x * (1.f + erff(x * 0.7071067811865475f)); }

union pk2 { bf16 h[2]; unsigned int u; };

// =====================================================================
// Generic bf16 WMMA GEMM:  C[z] = A[z] * B[z] (+bias[n]) (+C) (*scale+shift)
// A: M x K (row stride lda, batch stride sAz, rows offset by row_start)
// B: K x N (row stride ldb, batch stride sBz)
// Tiles: BM=64, BN=64, BK=32; 128 threads = 4 waves; wave w owns rows [16w,16w+16)
//
// LDS layouts are fragment-friendly:
//   As : row-major, K permuted [0..7,16..23,8..15,24..31] -> lane A-frag is
//        32 contiguous bytes at &As[row*40 + 16*g]
//   Bst: transposed [col][k] -> lane B-frag is 16 bytes at &Bst[col*40 + 8*g]
//        plus 16 bytes at &Bst[col*40 + 16 + 8*g]
// =====================================================================
__global__ __launch_bounds__(128) void k_gemm(
    const bf16* __restrict__ A, long long sAz, int lda, int row_start,
    const bf16* __restrict__ Bm, long long sBz, int ldb,
    float* __restrict__ C, long long sCz, int ldc,
    const float* __restrict__ bias, int addC,
    const float* __restrict__ scale, const float* __restrict__ shift, int ss_stride,
    int M, int N, int K)
{
  __shared__ bf16 As[64 * 40];
  __shared__ bf16 Bst[64 * 40];

  const int tid  = threadIdx.x;
  const int wv   = tid >> 5;
  const int lane = tid & 31;
  const int r16  = lane & 15;
  const int g    = lane >> 4;
  const int m0   = blockIdx.y * 64;
  const int n0   = blockIdx.x * 64;
  const bf16* Ab = A  + (long long)blockIdx.z * sAz;
  const bf16* Bb = Bm + (long long)blockIdx.z * sBz;

  v8f acc[4] = {};

  const int ksteps = (K + 31) >> 5;
  for (int ks = 0; ks < ksteps; ++ks) {
    const int k0 = ks << 5;

    // ---- stage A (64x32, permuted K): each thread = one half-row (16 elems) ----
    {
      const int row = tid >> 1;
      const int cb  = (tid & 1) << 4;          // 0 or 16
      const int gr  = m0 + row;
      bf16* dstrow = &As[row * 40];
      const bf16* src = Ab + (long long)(row_start + gr) * lda + (k0 + cb);
      const bool full = (gr < M) && (k0 + cb + 16 <= K);
      if (full && ((((uintptr_t)src) & 15) == 0)) {
        // K cb..cb+7 and cb+8..cb+15 land at permuted offsets
        const int d0 = (cb == 0) ? 0  : 8;
        const int d1 = (cb == 0) ? 16 : 24;
        async_cp16(src,     dstrow + d0);
        async_cp16(src + 8, dstrow + d1);
      } else {
#pragma unroll
        for (int e = 0; e < 16; ++e) {
          const int k  = cb + e;
          const int kk = k0 + k;
          const bf16 v = (gr < M && kk < K) ? src[e] : (bf16)0.f;
          const int pos = (k < 8) ? k : (k < 16) ? (k + 8) : (k < 24) ? (k - 8) : k;
          dstrow[pos] = v;
        }
      }
    }

    // ---- stage B (32x64) transposed into Bst[col][k], K-pairs packed b32 ----
    {
      const int krp = tid >> 3;                // 0..15 -> rows 2*krp, 2*krp+1
      const int cc  = (tid & 7) << 3;          // 0..56
      const int gk0 = k0 + 2 * krp;
      const int gn  = n0 + cc;
      const bf16* s0 = Bb + (long long)gk0 * ldb + gn;
      const bf16* s1 = s0 + ldb;
      const bool fast = (gk0 + 1 < K) && (gn + 8 <= N) &&
                        ((((uintptr_t)s0) & 15) == 0) && ((((uintptr_t)s1) & 15) == 0);
      if (fast) {
        bf16x8 r0 = *(const bf16x8*)s0;
        bf16x8 r1 = *(const bf16x8*)s1;
#pragma unroll
        for (int e = 0; e < 8; ++e) {
          pk2 p; p.h[0] = r0[e]; p.h[1] = r1[e];
          *(unsigned int*)&Bst[(cc + e) * 40 + 2 * krp] = p.u;
        }
      } else {
#pragma unroll
        for (int e = 0; e < 8; ++e) {
          const bool cok = (gn + e) < N;
          pk2 p;
          p.h[0] = (gk0     < K && cok) ? s0[e] : (bf16)0.f;
          p.h[1] = (gk0 + 1 < K && cok) ? s1[e] : (bf16)0.f;
          *(unsigned int*)&Bst[(cc + e) * 40 + 2 * krp] = p.u;
        }
      }
    }

    if (ks + 1 < ksteps) {  // global_prefetch_b8 of next B K-slab
      __builtin_prefetch(Bb + (long long)(k0 + 32) * ldb + n0 + (tid & 7) * 8, 0, 1);
    }
    wait_stage_barrier();

    // ---- fragments: 2x ds_load_b128 each ----
    v16bf a;
    {
      const bf16* ap = &As[(wv * 16 + r16) * 40 + 16 * g];
      bf16x8 lo = *(const bf16x8*)ap;
      bf16x8 hi = *(const bf16x8*)(ap + 8);
#pragma unroll
      for (int i = 0; i < 8; ++i) { a[i] = lo[i]; a[8 + i] = hi[i]; }
    }
    v16bf bf[4];
#pragma unroll
    for (int nt = 0; nt < 4; ++nt) {
      const bf16* bp = &Bst[(nt * 16 + r16) * 40 + 8 * g];
      bf16x8 lo = *(const bf16x8*)bp;
      bf16x8 hi = *(const bf16x8*)(bp + 16);
#pragma unroll
      for (int i = 0; i < 8; ++i) { bf[nt][i] = lo[i]; bf[nt][8 + i] = hi[i]; }
    }
#pragma unroll
    for (int nt = 0; nt < 4; ++nt) {
      acc[nt] = __builtin_amdgcn_wmma_f32_16x16x32_bf16(
          false, a, false, bf[nt], (short)0, acc[nt], false, false);
    }
    __syncthreads();
  }

  float* Cb = C + (long long)blockIdx.z * sCz;
#pragma unroll
  for (int nt = 0; nt < 4; ++nt) {
#pragma unroll
    for (int i = 0; i < 8; ++i) {
      const int gm = m0 + wv * 16 + i + (g << 3);
      const int gn = n0 + nt * 16 + r16;
      if (gm < M && gn < N) {
        const long long off = (long long)gm * ldc + gn;
        float v = acc[nt][i];
        if (bias)  v += bias[gn];
        if (addC)  v += Cb[off];
        if (scale) {
          const long long so = (long long)blockIdx.z * ss_stride + gn;
          v = v * scale[so] + shift[so];
        }
        Cb[off] = v;
      }
    }
  }
}

// =====================================================================
// Elementwise / prep kernels
// =====================================================================
__global__ __launch_bounds__(256) void k_stats(const float* __restrict__ x,
                                               float* __restrict__ mean, float* __restrict__ stdev) {
  int t = blockIdx.x * 256 + threadIdx.x;
  if (t >= B_ * CIN_) return;
  int b = t / CIN_, c = t % CIN_;
  float s = 0.f, ss = 0.f;
  for (int l = 0; l < L_; ++l) {
    float v = x[((long long)b * L_ + l) * CIN_ + c];
    s += v; ss += v * v;
  }
  float m = s / (float)L_;
  float var = ss / (float)L_ - m * m;
  mean[t] = m;
  stdev[t] = sqrtf(fmaxf(var, 0.f));
}

// circular-padded normalized input: xp[b, 0..97, c], plus zero tail pad
__global__ __launch_bounds__(256) void k_xp(const float* __restrict__ x,
                                            const float* __restrict__ mean, const float* __restrict__ stdev,
                                            bf16* __restrict__ xp, int n_main, int n_total) {
  int i = blockIdx.x * 256 + threadIdx.x;
  if (i >= n_total) return;
  if (i >= n_main) { xp[i] = (bf16)0.f; return; }
  int c  = i % CIN_;
  int lp = (i / CIN_) % XPL_;
  int b  = i / (CIN_ * XPL_);
  int ls = (lp == 0) ? (L_ - 1) : (lp == XPL_ - 1) ? 0 : (lp - 1);
  float v = (x[((long long)b * L_ + ls) * CIN_ + c] - mean[b * CIN_ + c]) / stdev[b * CIN_ + c];
  xp[i] = (bf16)v;
}

__global__ __launch_bounds__(256) void k_cvt(const float* __restrict__ s, bf16* __restrict__ d,
                                             int n_src, int n_total) {
  int i = blockIdx.x * 256 + threadIdx.x;
  if (i >= n_total) return;
  d[i] = (i < n_src) ? (bf16)s[i] : (bf16)0.f;
}

// Whcat[i] : [512 x 1536] bf16 = [Wt_h | Wa_h | Wh_h] (bottom halves), both layers
__global__ __launch_bounds__(256) void k_whcat(const float* __restrict__ Wt, const float* __restrict__ Wa,
                                               const float* __restrict__ Wh, bf16* __restrict__ out) {
  int idx = blockIdx.x * 256 + threadIdx.x;
  if (idx >= NL_ * 512 * 1536) return;
  int n = idx % 1536;
  int r = (idx / 1536) % 512;
  int i = idx / (1536 * 512);
  const float* src; int col;
  if (n < 512)       { src = Wt; col = n; }
  else if (n < 1024) { src = Wa; col = n - 512; }
  else               { src = Wh; col = n - 1024; }
  out[idx] = (bf16)src[(((long long)i * 1024) + 512 + r) * 512 + col];
}

// enc[b,l,d] = x_mark @ temp_w + sinusoidal positional embedding
__global__ __launch_bounds__(256) void k_biaspe(const float* __restrict__ xmark,
                                                const float* __restrict__ tempw, float* __restrict__ enc) {
  int idx = blockIdx.x * 256 + threadIdx.x;
  if (idx >= B_ * L_ * D_) return;
  int d = idx % D_;
  int l = (idx / D_) % L_;
  int b = idx / (D_ * L_);
  int i2 = d & ~1;
  float div = expf(-(float)i2 * (9.210340371976184f / (float)D_));
  float arg = (float)l * div;
  float acc = (d & 1) ? cosf(arg) : sinf(arg);
  const float* xm = xmark + ((long long)b * L_ + l) * NMARK_;
#pragma unroll
  for (int m = 0; m < NMARK_; ++m) acc += xm[m] * tempw[m * D_ + d];
  enc[idx] = acc;
}

// enc_out += pl_b[t]; also produce bf16 copy (RNN layer-0 input)
__global__ __launch_bounds__(256) void k_rowbias(const float* __restrict__ plb,
                                                 float* __restrict__ eo, bf16* __restrict__ eob) {
  int i = blockIdx.x * 256 + threadIdx.x;
  if (i >= B_ * TOUT_ * D_) return;
  int t = (i / D_) % TOUT_;
  float v = eo[i] + plb[t];
  eo[i] = v; eob[i] = (bf16)v;
}

__global__ __launch_bounds__(256) void k_init_state(const float* __restrict__ h0, const float* __restrict__ ce0,
                                                    const float* __restrict__ ct0, float* __restrict__ h,
                                                    float* __restrict__ ce, float* __restrict__ ct,
                                                    bf16* __restrict__ hbf) {
  int i = blockIdx.x * 256 + threadIdx.x;
  if (i >= B_ * H_) return;
  float v = h0[i];
  h[i] = v; hbf[i] = (bf16)v; ce[i] = ce0[i]; ct[i] = ct0[i];
}

// step part 1: gate nonlinearities, ct/ce update, h_up, hw
__global__ __launch_bounds__(256) void k_step1(const float* __restrict__ Gx, const float* __restrict__ gates,
                                               const float* __restrict__ h, float* __restrict__ ce,
                                               float* __restrict__ ct, float* __restrict__ hw,
                                               bf16* __restrict__ hup, int t) {
  int idx = blockIdx.x * 256 + threadIdx.x;
  if (idx >= B_ * H_) return;
  int b = idx >> 9, j = idx & 511;
  const float* gx = Gx + ((long long)(b * TOUT_ + t)) * 2048;
  const float* gt = gates + (long long)b * 1536;
  float tpre = gt[j] + gx[j];
  float apre = gt[512 + j] + gx[512 + j];
  float hpre = gt[1024 + j] + gx[1024 + j];
  float ctv = ct[idx], cev = ce[idx], hv = h[idx];
  ctv = fmaxf(ctv + tanhf(tpre), 0.f) - 1.f;
  float leap = (ctv <= 0.f) ? 1.f : 0.f;
  ctv = (1.f - leap) * ctv;
  float energy = leap * cev;
  cev = fmaxf(cev - energy + sigmoidf_(apre), 0.f);
  ct[idx] = ctv; ce[idx] = cev;
  hw[idx] = sigmoidf_(hpre);
  hup[idx] = (bf16)(hv * energy);
}

// step part 2: hs, h update, write output sequence
__global__ __launch_bounds__(256) void k_step2(const float* __restrict__ Gx, const float* __restrict__ esum,
                                               float* __restrict__ h, const float* __restrict__ hw,
                                               bf16* __restrict__ hbf, float* __restrict__ curf,
                                               bf16* __restrict__ curbf, int t) {
  int idx = blockIdx.x * 256 + threadIdx.x;
  if (idx >= B_ * H_) return;
  int b = idx >> 9, j = idx & 511;
  const float* gx = Gx + ((long long)(b * TOUT_ + t)) * 2048;
  float hs = tanhf(gx[1536 + j] + esum[idx]);
  float w = hw[idx];
  float hn = tanhf((1.f - w) * h[idx] + w * hs);
  h[idx] = hn; hbf[idx] = (bf16)hn;
  long long o = ((long long)(b * TOUT_ + t)) * H_ + j;
  curf[o] = hn; curbf[o] = (bf16)hn;
}

// LayerNorm(512) + ReLU -> bf16; one block per row
__global__ __launch_bounds__(256) void k_ln_relu(const float* __restrict__ z, const float* __restrict__ g,
                                                 const float* __restrict__ b, bf16* __restrict__ out) {
  __shared__ float shs[256], shq[256];
  int row = blockIdx.x, tid = threadIdx.x;
  const float* zr = z + (long long)row * H_;
  float s = 0.f, ss = 0.f;
  for (int j = tid; j < H_; j += 256) { float v = zr[j]; s += v; ss += v * v; }
  shs[tid] = s; shq[tid] = ss; __syncthreads();
  for (int st = 128; st > 0; st >>= 1) {
    if (tid < st) { shs[tid] += shs[tid + st]; shq[tid] += shq[tid + st]; }
    __syncthreads();
  }
  float m = shs[0] / (float)H_;
  float var = shq[0] / (float)H_ - m * m;
  float rstd = rsqrtf(var + 1e-5f);
  for (int j = tid; j < H_; j += 256) {
    float v = (zr[j] - m) * rstd * g[j] + b[j];
    out[(long long)row * H_ + j] = (bf16)fmaxf(v, 0.f);
  }
}

// e = gelu(gelu(gelu(z2)) + enc_out)
__global__ __launch_bounds__(256) void k_gelu_res(const float* __restrict__ z2, const float* __restrict__ eo,
                                                  bf16* __restrict__ eb) {
  int i = blockIdx.x * 256 + threadIdx.x;
  if (i >= B_ * TOUT_ * D_) return;
  float v = geluf_(z2[i]);
  v = geluf_(v);
  v = geluf_(v + eo[i]);
  eb[i] = (bf16)v;
}

// =====================================================================
// Host orchestration
// =====================================================================
extern "C" void kernel_launch(void* const* d_in, const int* in_sizes, int n_in,
                              void* d_out, int out_size, void* d_ws, size_t ws_size,
                              hipStream_t stream) {
  (void)in_sizes; (void)n_in; (void)out_size; (void)ws_size;
  const float* x_enc  = (const float*)d_in[0];
  const float* x_mark = (const float*)d_in[1];
  const float* conv_w = (const float*)d_in[4];
  const float* temp_w = (const float*)d_in[5];
  const float* pl_w   = (const float*)d_in[6];
  const float* pl_b   = (const float*)d_in[7];
  const float* Wt     = (const float*)d_in[8];
  const float* bt     = (const float*)d_in[9];
  const float* Wa     = (const float*)d_in[10];
  const float* ba     = (const float*)d_in[11];
  const float* We     = (const float*)d_in[12];
  const float* be     = (const float*)d_in[13];
  const float* Wh     = (const float*)d_in[14];
  const float* bh     = (const float*)d_in[15];
  const float* ow1_w  = (const float*)d_in[16];
  const float* ow1_b  = (const float*)d_in[17];
  const float* ln_g   = (const float*)d_in[18];
  const float* ln_b   = (const float*)d_in[19];
  const float* ow2_w  = (const float*)d_in[20];
  const float* ow2_b  = (const float*)d_in[21];
  const float* proj_w = (const float*)d_in[22];
  const float* proj_b = (const float*)d_in[23];
  const float* h0     = (const float*)d_in[24];
  const float* ce0    = (const float*)d_in[25];
  const float* ct0    = (const float*)d_in[26];
  float* out = (float*)d_out;

  // ---- workspace layout ----
  size_t off = 0;
  auto alloc = [&](size_t bytes) -> void* {
    void* p = (char*)d_ws + off;
    off = (off + bytes + 255) & ~(size_t)255;
    return p;
  };
  const long long nBLD = (long long)B_ * L_ * D_;       // 12.58M
  const long long nBTD = (long long)B_ * TOUT_ * D_;    // 25.17M
  float* mean   = (float*)alloc((size_t)B_ * CIN_ * 4);
  float* stdev  = (float*)alloc((size_t)B_ * CIN_ * 4);
  bf16*  xp     = (bf16*) alloc(((size_t)B_ * XPL_ * CIN_ + 64) * 2);
  bf16*  convwb = (bf16*) alloc((size_t)KCONV_ * D_ * 2);
  bf16*  plwb   = (bf16*) alloc((size_t)TOUT_ * L_ * 2);
  bf16*  ow1b   = (bf16*) alloc((size_t)H_ * H_ * 2);
  bf16*  ow2b   = (bf16*) alloc((size_t)H_ * D_ * 2);
  bf16*  projwb = (bf16*) alloc((size_t)D_ * CIN_ * 2);
  bf16*  Wtb    = (bf16*) alloc((size_t)NL_ * 1024 * 512 * 2);
  bf16*  Wab    = (bf16*) alloc((size_t)NL_ * 1024 * 512 * 2);
  bf16*  Whb    = (bf16*) alloc((size_t)NL_ * 1024 * 512 * 2);
  bf16*  Web    = (bf16*) alloc((size_t)NL_ * 1024 * 512 * 2);
  bf16*  whcat  = (bf16*) alloc((size_t)NL_ * 512 * 1536 * 2);
  float* enc    = (float*)alloc((size_t)nBLD * 4);
  bf16*  encb   = (bf16*) alloc((size_t)nBLD * 2);
  float* encout = (float*)alloc((size_t)nBTD * 4);
  bf16*  encoutb= (bf16*) alloc((size_t)nBTD * 2);
  float* cur    = (float*)alloc((size_t)nBTD * 4);
  bf16*  curb   = (bf16*) alloc((size_t)nBTD * 2);
  float* Gx     = (float*)alloc((size_t)B_ * TOUT_ * 2048 * 4);   // 402 MB, reused for head
  float* gates  = (float*)alloc((size_t)B_ * 1536 * 4);
  float* esum   = (float*)alloc((size_t)B_ * H_ * 4);
  float* hbuf   = (float*)alloc((size_t)B_ * H_ * 4);
  float* cebuf  = (float*)alloc((size_t)B_ * H_ * 4);
  float* ctbuf  = (float*)alloc((size_t)B_ * H_ * 4);
  float* hwbuf  = (float*)alloc((size_t)B_ * H_ * 4);
  bf16*  hbfb   = (bf16*) alloc((size_t)B_ * H_ * 2);
  bf16*  hupb   = (bf16*) alloc((size_t)B_ * H_ * 2);
  // head buffers reuse the Gx region (free after RNN)
  float* z1     = Gx;
  bf16*  z1nb   = (bf16*)((char*)Gx + (size_t)100663296);
  float* z2     = (float*)((char*)Gx + (size_t)150994944);
  bf16*  ebf    = (bf16*)((char*)Gx + (size_t)251658240);

  auto nb = [](long long n) { return dim3((unsigned)((n + 255) / 256)); };
  auto gemm = [&](const bf16* A, long long sAz, int lda, int rs,
                  const bf16* Bm, long long sBz, int ldb,
                  float* C, long long sCz, int ldc,
                  const float* bias, int addC,
                  const float* scale, const float* shift, int sst,
                  int M, int N, int K, int batch) {
    dim3 g((N + 63) / 64, (M + 63) / 64, batch);
    k_gemm<<<g, 128, 0, stream>>>(A, sAz, lda, rs, Bm, sBz, ldb, C, sCz, ldc,
                                  bias, addC, scale, shift, sst, M, N, K);
  };

  // ---- prep ----
  k_stats<<<nb((long long)B_ * CIN_), 256, 0, stream>>>(x_enc, mean, stdev);
  k_xp<<<nb((long long)B_ * XPL_ * CIN_ + 64), 256, 0, stream>>>(
      x_enc, mean, stdev, xp, B_ * XPL_ * CIN_, B_ * XPL_ * CIN_ + 64);
  k_cvt<<<nb((long long)KCONV_ * D_), 256, 0, stream>>>(conv_w, convwb, 963 * D_, KCONV_ * D_);
  k_cvt<<<nb((long long)TOUT_ * L_), 256, 0, stream>>>(pl_w, plwb, TOUT_ * L_, TOUT_ * L_);
  k_cvt<<<nb((long long)H_ * H_), 256, 0, stream>>>(ow1_w, ow1b, H_ * H_, H_ * H_);
  k_cvt<<<nb((long long)H_ * D_), 256, 0, stream>>>(ow2_w, ow2b, H_ * D_, H_ * D_);
  k_cvt<<<nb((long long)D_ * CIN_), 256, 0, stream>>>(proj_w, projwb, D_ * CIN_, D_ * CIN_);
  const int nW = NL_ * 1024 * 512;
  k_cvt<<<nb(nW), 256, 0, stream>>>(Wt, Wtb, nW, nW);
  k_cvt<<<nb(nW), 256, 0, stream>>>(Wa, Wab, nW, nW);
  k_cvt<<<nb(nW), 256, 0, stream>>>(Wh, Whb, nW, nW);
  k_cvt<<<nb(nW), 256, 0, stream>>>(We, Web, nW, nW);
  k_whcat<<<nb((long long)NL_ * 512 * 1536), 256, 0, stream>>>(Wt, Wa, Wh, whcat);

  // ---- embedding: enc = conv1d(xn) + mark@temp_w + pos ----
  k_biaspe<<<nb(nBLD), 256, 0, stream>>>(x_mark, temp_w, enc);
  gemm(xp, (long long)XPL_ * CIN_, CIN_, 0, convwb, 0, D_, enc, (long long)L_ * D_, D_,
       nullptr, 1, nullptr, nullptr, 0, L_, D_, KCONV_, B_);
  k_cvt<<<nb(nBLD), 256, 0, stream>>>(enc, encb, (int)nBLD, (int)nBLD);

  // ---- predict_linear: enc_out[b] = pl_w @ enc[b] + pl_b ----
  gemm(plwb, 0, L_, 0, encb, (long long)L_ * D_, D_, encout, (long long)TOUT_ * D_, D_,
       nullptr, 0, nullptr, nullptr, 0, TOUT_, D_, L_, B_);
  k_rowbias<<<nb(nBTD), 256, 0, stream>>>(pl_b, encout, encoutb);

  // ---- stacked QURNN layers ----
  const int MBT = B_ * TOUT_;     // 49152
  for (int i = 0; i < NL_; ++i) {
    k_init_state<<<nb((long long)B_ * H_), 256, 0, stream>>>(
        h0 + (long long)i * B_ * H_, ce0 + (long long)i * B_ * H_, ct0 + (long long)i * B_ * H_,
        hbuf, cebuf, ctbuf, hbfb);
    const bf16* xin = (i == 0) ? encoutb : curb;
    const bf16* Wfull[4] = { Wtb + (long long)i * 524288, Wab + (long long)i * 524288,
                             Whb + (long long)i * 524288, Web + (long long)i * 524288 };
    const float* biases[4] = { bt + i * H_, ba + i * H_, bh + i * H_, be + i * H_ };
    // precompute x-projections (+bias) for all timesteps: Gx[(b,t), 4x512]
    for (int s = 0; s < 4; ++s)
      gemm(xin, 0, D_, 0, Wfull[s], 0, H_, Gx + s * 512, 0, 2048,
           biases[s], 0, nullptr, nullptr, 0, MBT, H_, D_, 1);
    // sequential scan: only the h-dependent GEMMs per step
    for (int t = 0; t < TOUT_; ++t) {
      gemm(hbfb, 0, H_, 0, whcat + (long long)i * 512 * 1536, 0, 1536, gates, 0, 1536,
           nullptr, 0, nullptr, nullptr, 0, B_, 1536, H_, 1);
      k_step1<<<nb((long long)B_ * H_), 256, 0, stream>>>(Gx, gates, hbuf, cebuf, ctbuf, hwbuf, hupb, t);
      gemm(hupb, 0, H_, 0, Web + (long long)i * 524288 + 262144, 0, H_, esum, 0, H_,
           nullptr, 0, nullptr, nullptr, 0, B_, H_, H_, 1);
      k_step2<<<nb((long long)B_ * H_), 256, 0, stream>>>(Gx, esum, hbuf, hwbuf, hbfb, cur, curb, t);
    }
  }

  // ---- output head ----
  gemm(curb, 0, H_, 0, ow1b, 0, H_, z1, 0, H_, ow1_b, 0, nullptr, nullptr, 0, MBT, H_, H_, 1);
  k_ln_relu<<<dim3(MBT), 256, 0, stream>>>(z1, ln_g, ln_b, z1nb);
  gemm(z1nb, 0, H_, 0, ow2b, 0, D_, z2, 0, D_, ow2_b, 0, nullptr, nullptr, 0, MBT, D_, H_, 1);
  k_gelu_res<<<nb(nBTD), 256, 0, stream>>>(z2, encout, ebf);
  // final projection of the last PRED_ timesteps, fused de-normalization
  gemm(ebf, (long long)TOUT_ * D_, D_, TOUT_ - PRED_, projwb, 0, CIN_,
       out, (long long)PRED_ * CIN_, CIN_, proj_b, 0, stdev, mean, CIN_,
       PRED_, CIN_, D_, B_);
}
